// D_86036784873832
// MI455X (gfx1250) — matrix operand order repaired and checked by reading, not compiled, not run on hardware
//
#include <hip/hip_runtime.h>
#include <hip/hip_bf16.h>

typedef _Float16 half_t;
typedef __attribute__((ext_vector_type(16))) _Float16 v16h;
typedef __attribute__((ext_vector_type(8)))  _Float16 v8h;
typedef __attribute__((ext_vector_type(8)))  float    v8f;

// ---------------------------------------------------------------------------
// WMMA helpers (CDNA5 wave32, V_WMMA_F32_16X16X32_F16)
// ---------------------------------------------------------------------------
__device__ __forceinline__ v8f wmma_f16(v16h a, v16h b, v8f c) {
  return __builtin_amdgcn_wmma_f32_16x16x32_f16(false, a, false, b, (short)0, c,
                                                false, false);
}

// A-tile (16 M x 32 K) from row-major A, leading dim lda (elements).
// Layout: lanes 0-15 row M=lane hold K {0..15}; lanes 16-31 hold
// K {16..31}.  Two 16B vector loads per lane.
__device__ __forceinline__ v16h load_a16(const half_t* __restrict__ A, int lda) {
  int lane = threadIdx.x & 31;
  const half_t* p = A + (size_t)(lane & 15) * lda + ((lane & 16) ? 16 : 0);
  v8h lo = *(const v8h*)p;
  v8h hi = *(const v8h*)(p + 8);
  v16h r;
#pragma unroll
  for (int e = 0; e < 8; ++e) { r[e] = lo[e]; r[8 + e] = hi[e]; }
  return r;
}

// B-tile (32 K x 16 N) from TRANSPOSED source Bt[N, K] row-major (ld = ldb).
// Lane holding column N gets its 16 K-values contiguously: two 16B loads.
__device__ __forceinline__ v16h load_b16t(const half_t* __restrict__ Bt, int ldb) {
  int lane = threadIdx.x & 31;
  const half_t* p = Bt + (size_t)(lane & 15) * ldb + ((lane & 16) ? 16 : 0);
  v8h lo = *(const v8h*)p;
  v8h hi = *(const v8h*)(p + 8);
  v16h r;
#pragma unroll
  for (int e = 0; e < 8; ++e) { r[e] = lo[e]; r[8 + e] = hi[e]; }
  return r;
}

// ---------------------------------------------------------------------------
// Sliding-window mean / population std.
// Output xT = [mean ; std] transposed: [n, P, 512] (f32 + f16), pad rows zeroed.
// ---------------------------------------------------------------------------
__global__ void winstats_kernel(const float* __restrict__ feat,
                                float* __restrict__ xT_f32,
                                half_t* __restrict__ xT_h,
                                int K, int s, int h, int w, int hw, int P) {
  int idx = blockIdx.x * blockDim.x + threadIdx.x;
  int total = 2 * 256 * P;
  if (idx >= total) return;
  int p = idx % P;
  int c = (idx / P) % 256;
  int n = idx / (P * 256);
  float mean = 0.f, stdv = 0.f;
  if (p < hw) {
    int oy = p / w, ox = p % w;
    const float* src = feat + (((size_t)n * 256 + c) * 48 + oy * s) * 96 + ox * s;
    float sm = 0.f, sq = 0.f;
    for (int yy = 0; yy < K; ++yy) {
      const float* row = src + yy * 96;
      for (int xx = 0; xx < K; ++xx) {
        float v = row[xx];
        sm += v;
        sq += v * v;
      }
    }
    float inv = 1.0f / (float)(K * K);
    mean = sm * inv;
    float var = sq * inv - mean * mean;
    if (var < 0.f) var = 0.f;
    stdv = sqrtf(var + 1e-12f);
  }
  size_t base = ((size_t)n * P + p) * 512;
  xT_f32[base + c] = mean;
  xT_f32[base + 256 + c] = stdv;
  xT_h[base + c] = (half_t)mean;
  xT_h[base + 256 + c] = (half_t)stdv;
}

// ---------------------------------------------------------------------------
// WMMA GEMM, 4 waves / block, macro-tile 64M x 16N.
// C[M,N] = A[M,K](f16 row-major) @ B (activations, TRANSPOSED [N,K] f16).
// The shared 16N x 32K B panel is double-buffered through LDS with CDNA5
// async global->LDS copies (global_load_async_to_lds_b128 / s_wait_asynccnt),
// reused by all 4 waves. Epilogue fuses bias / BN / residual / ReLU
// (template-selected, branch-free) and stores transposed [N,M] (packed 16B)
// or row-major [M,N] (TOUT=false, for the attention V operand).
// ---------------------------------------------------------------------------
template <bool BIAS, bool BN, bool RES, bool RELU, bool TOUT>
__global__ void __launch_bounds__(128)
gemm16t_kernel(const half_t* __restrict__ A, const half_t* __restrict__ Bt,
               const float* __restrict__ bias, const float* __restrict__ bnscale,
               const float* __restrict__ bnshift, const float* __restrict__ residT,
               half_t* __restrict__ C,
               int M, int N, int K,
               size_t strideB, size_t strideC, size_t strideR) {
  __shared__ half_t Bl[2][16 * 32];
  int tid = threadIdx.x;
  int wave = tid >> 5;
  int lane = tid & 31;
  int n0 = blockIdx.x * 16;
  int m0 = blockIdx.y * 64 + wave * 16;
  int batch = blockIdx.z;
  const half_t* Bp = Bt + (size_t)batch * strideB + (size_t)n0 * K;

  // prologue: stage k=0 panel into buffer 0 (64 async b128 chunks)
  if (tid < 64) {
    int row = tid >> 2, part = (tid & 3) * 8;
    asm volatile("global_load_async_to_lds_b128 %0, %1, off" ::"v"(
                     (uint32_t)(uintptr_t)&Bl[0][row * 32 + part]),
                 "v"(Bp + (size_t)row * K + part)
                 : "memory");
  }
  v8f acc = {};
  int nk = K >> 5;
  for (int i = 0; i < nk; ++i) {
    int cur = i & 1;
    asm volatile("s_wait_asynccnt 0" ::: "memory");
    __syncthreads();  // buf[cur] visible to all waves; prev buf free
    if (i + 1 < nk && tid < 64) {  // kick next panel while computing
      int row = tid >> 2, part = (tid & 3) * 8;
      asm volatile("global_load_async_to_lds_b128 %0, %1, off" ::"v"(
                       (uint32_t)(uintptr_t)&Bl[cur ^ 1][row * 32 + part]),
                   "v"(Bp + (size_t)row * K + (i + 1) * 32 + part)
                   : "memory");
    }
    int k = i * 32;
    if (k + 32 < K) __builtin_prefetch(A + (size_t)m0 * K + k + 32, 0, 1);
    v16h a = load_a16(A + (size_t)m0 * K + k, K);
    v16h b;
    {
      const half_t* p = &Bl[cur][(lane & 15) * 32 + ((lane & 16) ? 16 : 0)];
      v8h lo = *(const v8h*)p;
      v8h hi = *(const v8h*)(p + 8);
#pragma unroll
      for (int e = 0; e < 8; ++e) { b[e] = lo[e]; b[8 + e] = hi[e]; }
    }
    acc = wmma_f16(a, b, acc);
  }

  int col = lane & 15;
  int rowoff = (lane & 16) ? 8 : 0;
  const float* rp =
      RES ? residT + (size_t)batch * strideR + (size_t)(n0 + col) * M + m0 + rowoff
          : nullptr;
  v8h outh;
#pragma unroll
  for (int r = 0; r < 8; ++r) {
    int m = m0 + rowoff + r;
    float v = acc[r];
    if (BIAS) v += bias[m];
    if (BN)   v = v * bnscale[m] + bnshift[m];
    if (RES)  v += rp[r];
    if (RELU) v = fmaxf(v, 0.0f);
    if (TOUT) outh[r] = (half_t)v;
    else      C[(size_t)batch * strideC + (size_t)m * N + n0 + col] = (half_t)v;
  }
  if (TOUT)
    *(v8h*)(C + (size_t)batch * strideC + (size_t)(n0 + col) * M + m0 + rowoff) = outh;
}

// ---------------------------------------------------------------------------
// Flash attention:  y[c,i] = sum_j softmax_j( scale * <Q_i, K_j> ) V[c,j]
// Q = txT [P,128], K = pxT [P,128] (transposed => all contiguous 16B loads),
// V = gx [128,P]. One wave owns 16 query rows, streams keys/values in blocks
// of 32 with an online softmax; hw..P-1 columns masked to -inf.
// Output yT [P,128].
// ---------------------------------------------------------------------------
__global__ void __launch_bounds__(32)
attn_kernel(const half_t* __restrict__ txT, const half_t* __restrict__ pxT,
            const half_t* __restrict__ gxv, half_t* __restrict__ yT,
            int hw, int P, float scale) {
  __shared__ half_t Pl[16][32];
  int lane = threadIdx.x & 31;
  int col = lane & 15;
  int rowoff = (lane & 16) ? 8 : 0;
  int i0 = blockIdx.x * 16;
  int nb = blockIdx.y;
  const half_t* Q  = txT + (size_t)nb * P * 128;
  const half_t* Kp = pxT + (size_t)nb * P * 128;
  const half_t* V  = gxv + (size_t)nb * 128 * P;
  half_t* Yt       = yT  + (size_t)nb * P * 128;

  // Q block: 4 A-tiles over the 128-deep contraction
  v16h qa[4];
#pragma unroll
  for (int cc = 0; cc < 4; ++cc)
    qa[cc] = load_a16(Q + (size_t)i0 * 128 + cc * 32, 128);

  v8f zero = {};
  v8f O[8];
#pragma unroll
  for (int t = 0; t < 8; ++t) O[t] = zero;
  float mrow[8], lrow[8];
#pragma unroll
  for (int r = 0; r < 8; ++r) { mrow[r] = -3.0e38f; lrow[r] = 0.f; }

  for (int j0 = 0; j0 < P; j0 += 32) {
    // S = Q K^T for 32 key columns (two 16x16 tiles)
    v8f S0 = zero, S1 = zero;
#pragma unroll
    for (int cc = 0; cc < 4; ++cc) {
      v16h b0 = load_b16t(Kp + (size_t)j0 * 128 + cc * 32, 128);
      v16h b1 = load_b16t(Kp + (size_t)(j0 + 16) * 128 + cc * 32, 128);
      S0 = wmma_f16(qa[cc], b0, S0);
      S1 = wmma_f16(qa[cc], b1, S1);
    }
    int c0g = j0 + col, c1g = j0 + 16 + col;
    float p0[8], p1[8], mx[8];
#pragma unroll
    for (int r = 0; r < 8; ++r) {
      float s0 = S0[r] * scale; if (c0g >= hw) s0 = -3.0e38f;
      float s1 = S1[r] * scale; if (c1g >= hw) s1 = -3.0e38f;
      p0[r] = s0; p1[r] = s1;
      mx[r] = fmaxf(s0, s1);
    }
    // row max across the 16 lanes of each half-wave (rows stay per-VGPR slot)
#pragma unroll
    for (int r = 0; r < 8; ++r) {
#pragma unroll
      for (int d = 1; d < 16; d <<= 1) mx[r] = fmaxf(mx[r], __shfl_xor(mx[r], d, 32));
    }
    float corr[8];
#pragma unroll
    for (int r = 0; r < 8; ++r) {
      float nm = fmaxf(mrow[r], mx[r]);
      corr[r] = __expf(mrow[r] - nm);
      mrow[r] = nm;
      p0[r] = __expf(p0[r] - nm);
      p1[r] = __expf(p1[r] - nm);
      float sm = p0[r] + p1[r];
#pragma unroll
      for (int d = 1; d < 16; d <<= 1) sm += __shfl_xor(sm, d, 32);
      lrow[r] = lrow[r] * corr[r] + sm;
    }
#pragma unroll
    for (int t = 0; t < 8; ++t)
#pragma unroll
      for (int r = 0; r < 8; ++r) O[t][r] *= corr[r];

    // stage P (16x32 f16) through LDS: C-layout -> A-layout transpose
    __syncthreads();
#pragma unroll
    for (int r = 0; r < 8; ++r) {
      Pl[rowoff + r][col]      = (half_t)p0[r];
      Pl[rowoff + r][16 + col] = (half_t)p1[r];
    }
    __syncthreads();
    v16h pa;
    {
      int row = lane & 15;
      int kh = (lane & 16) ? 16 : 0;
      v8h lo = *(const v8h*)&Pl[row][kh];
      v8h hi = *(const v8h*)&Pl[row][8 + kh];
#pragma unroll
      for (int e = 0; e < 8; ++e) { pa[e] = lo[e]; pa[8 + e] = hi[e]; }
    }
    // O += P @ V  (V columns contiguous along j -> 16B vector loads)
#pragma unroll
    for (int t = 0; t < 8; ++t) {
      const half_t* q = V + (size_t)(t * 16 + col) * P + j0 + ((lane & 16) ? 16 : 0);
      v8h b0 = *(const v8h*)q;
      v8h b1 = *(const v8h*)(q + 8);
      v16h bv;
#pragma unroll
      for (int e = 0; e < 8; ++e) { bv[e] = b0[e]; bv[8 + e] = b1[e]; }
      O[t] = wmma_f16(pa, bv, O[t]);
    }
  }
  float rl[8];
#pragma unroll
  for (int r = 0; r < 8; ++r) rl[r] = 1.0f / lrow[r];
#pragma unroll
  for (int t = 0; t < 8; ++t)
#pragma unroll
    for (int r = 0; r < 8; ++r) {
      float v = O[t][r] * rl[r];
      Yt[(size_t)(i0 + rowoff + r) * 128 + t * 16 + col] = (half_t)v;
    }
}

// ---------------------------------------------------------------------------
// Small helper kernels
// ---------------------------------------------------------------------------
__global__ void f2h_kernel(const float* __restrict__ s, half_t* __restrict__ d, int n) {
  int i = blockIdx.x * blockDim.x + threadIdx.x;
  if (i < n) d[i] = (half_t)s[i];
}

// bn params [nb, 4, C] -> inv = g*rsqrt(v+eps), shift = b - m*inv  ([nb, C])
__global__ void bnprep_kernel(const float* __restrict__ bn, float* __restrict__ inv,
                              float* __restrict__ shift, int C, int nb) {
  int idx = blockIdx.x * blockDim.x + threadIdx.x;
  if (idx >= C * nb) return;
  int i = idx / C, c = idx % C;
  const float* p = bn + (size_t)i * 4 * C;
  float g = p[c], b = p[C + c], m = p[2 * C + c], v = p[3 * C + c];
  float iv = g * rsqrtf(v + 1e-5f);
  inv[idx] = iv;
  shift[idx] = b - m * iv;
}

// logits from z3T [n,P,128]: out[(n*hw+p)*2+o] = 0.5*(b[o] + <w[o], z3T[p]>)
__global__ void final_kernel(const half_t* __restrict__ z3T, const float* __restrict__ m4w,
                             const float* __restrict__ m4b, float* __restrict__ out,
                             int hw, int P) {
  int idx = blockIdx.x * blockDim.x + threadIdx.x;
  if (idx >= 2 * hw * 2) return;
  int o = idx & 1;
  int t = idx >> 1;
  int p = t % hw;
  int n = t / hw;
  const half_t* z = z3T + ((size_t)n * P + p) * 128;
  const float* wrow = m4w + o * 128;
  float acc = m4b[o];
  for (int c = 0; c < 128; ++c) acc += wrow[c] * (float)z[c];
  out[((size_t)n * hw + p) * 2 + o] = acc * 0.5f;
}

// ---------------------------------------------------------------------------
// Host orchestration
// ---------------------------------------------------------------------------
extern "C" void kernel_launch(void* const* d_in, const int* in_sizes, int n_in,
                              void* d_out, int out_size, void* d_ws, size_t ws_size,
                              hipStream_t stream) {
  (void)in_sizes; (void)n_in; (void)out_size; (void)ws_size;
  const float* feature = (const float*)d_in[0];
  const float* g_w  = (const float*)d_in[1];
  const float* g_b  = (const float*)d_in[2];
  const float* th_w = (const float*)d_in[3];
  const float* th_b = (const float*)d_in[4];
  const float* ph_w = (const float*)d_in[5];
  const float* ph_b = (const float*)d_in[6];
  const float* o_w  = (const float*)d_in[7];
  const float* o_bn = (const float*)d_in[8];
  const float* m1_w = (const float*)d_in[9];
  const float* m1_bn = (const float*)d_in[10];
  const float* m2_w = (const float*)d_in[11];
  const float* m2_bn = (const float*)d_in[12];
  const float* m3_w = (const float*)d_in[13];
  const float* m3_bn = (const float*)d_in[14];
  const float* m4_w = (const float*)d_in[15];
  const float* m4_b = (const float*)d_in[16];
  float* out = (float*)d_out;

  // window configs: K = round(linspace(3,48,3)) = {3,26,48}
  const int Ks[3] = {3, 26, 48};
  const int Ss[3] = {1, 12, 23};
  const int Hs[3] = {46, 2, 1};
  const int Wd[3] = {94, 6, 3};
  const int outoff[3] = {0, 2 * 46 * 94 * 2, 2 * 46 * 94 * 2 + 2 * 2 * 6 * 2};
  const int PMAX = 4352;  // hw=4324 padded to multiple of 128

  size_t off = 0;
  char* base = (char*)d_ws;
  auto alloc = [&](size_t bytes) -> void* {
    off = (off + 255) & ~(size_t)255;
    void* p = base + off;
    off += bytes;
    return p;
  };
  half_t* wg  = (half_t*)alloc(sizeof(half_t) * 3 * 128 * 512);
  half_t* wth = (half_t*)alloc(sizeof(half_t) * 3 * 128 * 512);
  half_t* wph = (half_t*)alloc(sizeof(half_t) * 3 * 128 * 512);
  half_t* wo  = (half_t*)alloc(sizeof(half_t) * 3 * 512 * 128);
  half_t* wm1 = (half_t*)alloc(sizeof(half_t) * 3 * 256 * 512);
  half_t* wm2 = (half_t*)alloc(sizeof(half_t) * 3 * 128 * 256);
  half_t* wm3 = (half_t*)alloc(sizeof(half_t) * 3 * 128 * 128);
  float* o_inv  = (float*)alloc(sizeof(float) * 3 * 512);
  float* o_sh   = (float*)alloc(sizeof(float) * 3 * 512);
  float* m1_inv = (float*)alloc(sizeof(float) * 3 * 256);
  float* m1_sh  = (float*)alloc(sizeof(float) * 3 * 256);
  float* m2_inv = (float*)alloc(sizeof(float) * 3 * 128);
  float* m2_sh  = (float*)alloc(sizeof(float) * 3 * 128);
  float* m3_inv = (float*)alloc(sizeof(float) * 3 * 128);
  float* m3_sh  = (float*)alloc(sizeof(float) * 3 * 128);
  float*  xT_f32 = (float*)alloc(sizeof(float) * 2 * PMAX * 512);
  half_t* xT_h   = (half_t*)alloc(sizeof(half_t) * 2 * PMAX * 512);
  half_t* txT    = (half_t*)alloc(sizeof(half_t) * 2 * PMAX * 128);
  half_t* pxT    = (half_t*)alloc(sizeof(half_t) * 2 * PMAX * 128);
  half_t* gx     = (half_t*)alloc(sizeof(half_t) * 2 * 128 * PMAX);  // row-major (V)
  half_t* yT     = (half_t*)alloc(sizeof(half_t) * 2 * PMAX * 128);
  half_t* xoT    = (half_t*)alloc(sizeof(half_t) * 2 * PMAX * 512);
  half_t* z1T    = (half_t*)alloc(sizeof(half_t) * 2 * PMAX * 256);
  half_t* z2T    = (half_t*)alloc(sizeof(half_t) * 2 * PMAX * 128);
  half_t* z3T    = (half_t*)alloc(sizeof(half_t) * 2 * PMAX * 128);

  auto cvt = [&](const float* s, half_t* d, int n) {
    f2h_kernel<<<(n + 255) / 256, 256, 0, stream>>>(s, d, n);
  };
  cvt(g_w,  wg,  3 * 128 * 512);
  cvt(th_w, wth, 3 * 128 * 512);
  cvt(ph_w, wph, 3 * 128 * 512);
  cvt(o_w,  wo,  3 * 512 * 128);
  cvt(m1_w, wm1, 3 * 256 * 512);
  cvt(m2_w, wm2, 3 * 128 * 256);
  cvt(m3_w, wm3, 3 * 128 * 128);
  bnprep_kernel<<<(3 * 512 + 255) / 256, 256, 0, stream>>>(o_bn,  o_inv,  o_sh,  512, 3);
  bnprep_kernel<<<(3 * 256 + 255) / 256, 256, 0, stream>>>(m1_bn, m1_inv, m1_sh, 256, 3);
  bnprep_kernel<<<(3 * 128 + 255) / 256, 256, 0, stream>>>(m2_bn, m2_inv, m2_sh, 128, 3);
  bnprep_kernel<<<(3 * 128 + 255) / 256, 256, 0, stream>>>(m3_bn, m3_inv, m3_sh, 128, 3);

  const float attn_scale = 0.08838834764831845f;  // 1/sqrt(128)

  for (int i = 0; i < 3; ++i) {
    int K = Ks[i], s = Ss[i], h = Hs[i], w = Wd[i];
    int hw = h * w;
    int P = ((hw + 127) / 128) * 128;
    size_t sX = (size_t)P * 512;  // batch stride, 512-ch maps (transposed)
    size_t sC = (size_t)P * 128;  // batch stride, 128-ch maps
    size_t sZ1 = (size_t)P * 256;

    // 1. window statistics -> xT = [mean;std] (f32 + f16)
    int tot = 2 * 256 * P;
    winstats_kernel<<<(tot + 255) / 256, 256, 0, stream>>>(
        feature, xT_f32, xT_h, K, s, h, w, hw, P);

    // 2. Q/K/V 1x1 convs. Q,K stored transposed [P,128]; V row-major [128,P].
    dim3 g128(P / 16, 2, 2);  // M=128 -> 2 macro-tiles of 64
    gemm16t_kernel<true, false, false, false, true><<<g128, 128, 0, stream>>>(
        wth + (size_t)i * 128 * 512, xT_h, th_b + i * 128, nullptr, nullptr,
        nullptr, txT, 128, P, 512, sX, sC, 0);
    gemm16t_kernel<true, false, false, false, true><<<g128, 128, 0, stream>>>(
        wph + (size_t)i * 128 * 512, xT_h, ph_b + i * 128, nullptr, nullptr,
        nullptr, pxT, 128, P, 512, sX, sC, 0);
    gemm16t_kernel<true, false, false, false, false><<<g128, 128, 0, stream>>>(
        wg + (size_t)i * 128 * 512, xT_h, g_b + i * 128, nullptr, nullptr,
        nullptr, gx, 128, P, 512, sX, sC, 0);

    // 3. flash attention (online softmax, WMMA for QK^T and PV)
    attn_kernel<<<dim3(P / 16, 2), 32, 0, stream>>>(
        txT, pxT, gx, yT, hw, P, attn_scale);

    // 4. x = x + BN(o_w @ y)  (residual + BN fused)
    dim3 g512(P / 16, 8, 2);
    gemm16t_kernel<false, true, true, false, true><<<g512, 128, 0, stream>>>(
        wo + (size_t)i * 512 * 128, yT, nullptr, o_inv + i * 512, o_sh + i * 512,
        xT_f32, xoT, 512, P, 128, sC, sX, sX);

    // 5-7. MLP head with fused BN+ReLU
    dim3 g256(P / 16, 4, 2);
    gemm16t_kernel<false, true, false, true, true><<<g256, 128, 0, stream>>>(
        wm1 + (size_t)i * 256 * 512, xoT, nullptr, m1_inv + i * 256,
        m1_sh + i * 256, nullptr, z1T, 256, P, 512, sX, sZ1, 0);
    gemm16t_kernel<false, true, false, true, true><<<g128, 128, 0, stream>>>(
        wm2 + (size_t)i * 128 * 256, z1T, nullptr, m2_inv + i * 128,
        m2_sh + i * 128, nullptr, z2T, 128, P, 256, sZ1, sC, 0);
    gemm16t_kernel<false, true, false, true, true><<<g128, 128, 0, stream>>>(
        wm3 + (size_t)i * 128 * 128, z2T, nullptr, m3_inv + i * 128,
        m3_sh + i * 128, nullptr, z3T, 128, P, 128, sC, sC, 0);

    // 8. 2-channel logits, transposed to [n*h*w, 2]
    int tot4 = 2 * hw * 2;
    final_kernel<<<(tot4 + 255) / 256, 256, 0, stream>>>(
        z3T, m4_w + (size_t)i * 2 * 128, m4_b + i * 2, out + outoff[i], hw, P);
  }
}
